// PredictorNoEgo_39960375722582
// MI455X (gfx1250) — matrix operand order repaired and checked by reading, not compiled, not run on hardware
//
#include <hip/hip_runtime.h>
#include <hip/hip_bf16.h>

typedef __bf16 bf16;
typedef __attribute__((ext_vector_type(16))) __bf16 bf16x16;
typedef __attribute__((ext_vector_type(8)))  __bf16 bf16x8;
typedef __attribute__((ext_vector_type(8)))  float  f32x8;

#define B_SZ 8192
#define T_SZ 8
#define E_SZ 256
#define M_SZ 20
#define FS_SZ 12

__device__ __forceinline__ float sigm(float x){ return 1.f/(1.f+__expf(-x)); }
__device__ __forceinline__ float eluf(float x){ return x>0.f ? x : (__expf(x)-1.f); }

// ---- CDNA5 async global->LDS copy (ASYNCcnt-tracked), 16B per lane ----
// lds_off is the byte offset within the workgroup's (dynamic) LDS allocation.
__device__ __forceinline__ void async_g2l_b128(unsigned lds_off, const void* gptr){
  asm volatile("global_load_async_to_lds_b128 %0, %1, off"
               :: "v"(lds_off), "v"((unsigned long long)(uintptr_t)gptr)
               : "memory");
}
__device__ __forceinline__ void wait_async0(){
  asm volatile("s_wait_asynccnt 0x0" ::: "memory");
}

// ---- WMMA fragment loaders (CDNA5 16x16x32 bf16 layouts) ----
// A (16x32): lane L: M=L&15; element j: K = (j>>3)*16 + (L>>4)*8 + (j&7)
__device__ __forceinline__ bf16x16 ldfragA(const bf16* p, int half){
  bf16x8 lo = *(const bf16x8*)(p + half*8);
  bf16x8 hi = *(const bf16x8*)(p + 16 + half*8);
  bf16x16 r;
#pragma unroll
  for (int i=0;i<8;i++){ r[i]=lo[i]; r[i+8]=hi[i]; }
  return r;
}
// B (32x16): lane L: N=L&15; element j: K = (j>>3)*16 + (L>>4)*8 + (j&7)
__device__ __forceinline__ bf16x16 ldfragB(const bf16* p, int half){
  bf16x8 lo = *(const bf16x8*)(p + half*8);
  bf16x8 hi = *(const bf16x8*)(p + 16 + half*8);
  bf16x16 r;
#pragma unroll
  for (int i=0;i<8;i++){ r[i]=lo[i]; r[i+8]=hi[i]; }
  return r;
}

template<int NT>
__device__ __forceinline__ void zero_acc(f32x8* acc){
#pragma unroll
  for (int i=0;i<NT;i++){
#pragma unroll
    for (int r=0;r<8;r++) acc[i][r] = 0.f;
  }
}

// 16-row tile: C = A(16xK) @ W(NxK)^T  into acc[NT] col-tiles at n0
template<int NT>
__device__ __forceinline__ void gemm_acc(f32x8* acc, const bf16* __restrict__ A, int lda,
                                         const bf16* __restrict__ W, int ldw,
                                         int n0, int K, int lane, int nmax){
  const int row = lane & 15, half = lane >> 4;
  const bf16* arow = A + (size_t)row*lda;
  for (int kt = 0; kt < K; kt += 32){
    bf16x16 a = ldfragA(arow + kt, half);
#pragma unroll
    for (int i=0;i<NT;i++){
      int wr = n0 + i*16 + row;
      if (wr > nmax) wr = nmax;
      bf16x16 b = ldfragB(W + (size_t)wr*ldw + kt, half);
      acc[i] = __builtin_amdgcn_wmma_f32_16x16x32_bf16(false, a, false, b,
                                                       (short)0, acc[i], false, false);
    }
  }
}

// 32-row tile: two row-tiles share each B fragment (2 WMMAs per B load)
template<int NT>
__device__ __forceinline__ void gemm2_acc(f32x8 acc[2][NT], const bf16* __restrict__ A, int lda,
                                          const bf16* __restrict__ W, int ldw,
                                          int n0, int K, int lane){
  const int row = lane & 15, half = lane >> 4;
  const bf16* arow0 = A + (size_t)row*lda;
  const bf16* arow1 = A + (size_t)(16 + row)*lda;
  for (int kt = 0; kt < K; kt += 32){
    bf16x16 a0 = ldfragA(arow0 + kt, half);
    bf16x16 a1 = ldfragA(arow1 + kt, half);
#pragma unroll
    for (int i=0;i<NT;i++){
      bf16x16 b = ldfragB(W + (size_t)(n0 + i*16 + row)*ldw + kt, half);
      acc[0][i] = __builtin_amdgcn_wmma_f32_16x16x32_bf16(false, a0, false, b,
                                                          (short)0, acc[0][i], false, false);
      acc[1][i] = __builtin_amdgcn_wmma_f32_16x16x32_bf16(false, a1, false, b,
                                                          (short)0, acc[1][i], false, false);
    }
  }
}

// C/D layout: element r of lane: M = (lane>>4)*8 + r, N = lane&15 (within tile)
template<int NT, int ACT>  // ACT: 0 none, 1 relu, 2 elu
__device__ __forceinline__ void store_bf16(const f32x8* acc, bf16* dst, int ldn, int n0,
                                           const float* __restrict__ bias, int lane){
  const int col = lane & 15, mb = (lane >> 4) * 8;
#pragma unroll
  for (int i=0;i<NT;i++){
    int n = n0 + i*16 + col;
    float bv = bias[n];
#pragma unroll
    for (int r=0;r<8;r++){
      float v = acc[i][r] + bv;
      if (ACT==1) v = fmaxf(v, 0.f);
      else if (ACT==2) v = eluf(v);
      dst[(mb + r)*ldn + n] = (bf16)v;
    }
  }
}

template<int NT>
__device__ __forceinline__ void store_f32(const f32x8* acc, float* dst, int ldn, int n0,
                                          const float* __restrict__ bias, int lane){
  const int col = lane & 15, mb = (lane >> 4) * 8;
#pragma unroll
  for (int i=0;i<NT;i++){
    int n = n0 + i*16 + col;
    float bv = bias[n];
#pragma unroll
    for (int r=0;r<8;r++) dst[(mb + r)*ldn + n] = acc[i][r] + bv;
  }
}

__device__ __forceinline__ void ln_row(const float* __restrict__ x,
                                       const float* __restrict__ g,
                                       const float* __restrict__ b,
                                       bf16* __restrict__ y){
  float mu = 0.f;
  for (int j=0;j<E_SZ;j++) mu += x[j];
  mu *= (1.f/E_SZ);
  float var = 0.f;
  for (int j=0;j<E_SZ;j++){ float d = x[j]-mu; var += d*d; }
  var *= (1.f/E_SZ);
  float rs = rsqrtf(var + 1e-5f);
  for (int j=0;j<E_SZ;j++) y[j] = (bf16)((x[j]-mu)*rs*g[j] + b[j]);
}

// ---------------- weight conversion ----------------
__global__ __launch_bounds__(256) void cvt_bf16(const float* __restrict__ s,
                                                bf16* __restrict__ d, int n){
  int i = blockIdx.x*256 + threadIdx.x;
  if (i < n) d[i] = (bf16)s[i];
}
// extract v-slice (rows 2E..3E) of in_proj_w per mode: out is M x 256 x 256
__global__ __launch_bounds__(256) void cvt_vslice(const float* __restrict__ s,
                                                  bf16* __restrict__ d){
  int i = blockIdx.x*256 + threadIdx.x;
  int m = i >> 16;
  int r = i & 0xFFFF;
  d[i] = (bf16)s[(size_t)m*768*256 + 512*256 + r];
}

// ---------------- LSTM step (one layer, one timestep) ----------------
// 512 threads = 16 waves; 32-row batch tile; each wave owns 64 gate columns.
// LDS: At = 32x512 bf16 (32KB, async-staged A panel), G = 32x1024 f32 (128KB).
__global__ __launch_bounds__(512) void lstm_step(
    const bf16* __restrict__ hA, const bf16* __restrict__ hB,
    const bf16* __restrict__ w1, const bf16* __restrict__ w2,
    const float* __restrict__ b_ih, const float* __restrict__ b_hh,
    const float* __restrict__ w_ih0,     // non-null only for layer0 (1024x2 fp32)
    const float* __restrict__ obs, int t,
    float* __restrict__ c_state, bf16* __restrict__ h_out)
{
  extern __shared__ char smem[];
  bf16*  At = (bf16*)smem;                       // 32 x 512 (ld=512)
  float* G  = (float*)(smem + 32*512*2);         // 32 x 1024
  const int b0   = blockIdx.x * 32;
  const int tid  = threadIdx.x;
  const int lane = tid & 31, wave = tid >> 5;
  const int row  = lane & 15, half = lane >> 4;
  const int n0   = wave * 64;                    // 16 waves * 4 tiles * 16

  // async-stage A panel: hA rows -> cols [0,256); hB rows -> cols [256,512)
  // 32 rows x 512B (hA): 1024 chunks of 16B
  for (int i = tid; i < 1024; i += 512){
    int r = i >> 5, c = i & 31;                  // 32 chunks per row
    async_g2l_b128((unsigned)(r*1024 + c*16),
                   (const char*)(hA + (size_t)(b0 + r)*E_SZ) + c*16);
  }
  if (hB){
    for (int i = tid; i < 1024; i += 512){
      int r = i >> 5, c = i & 31;
      async_g2l_b128((unsigned)(r*1024 + 512 + c*16),
                     (const char*)(hB + (size_t)(b0 + r)*E_SZ) + c*16);
    }
  }
  wait_async0();
  __syncthreads();

  f32x8 acc[2][4];
  zero_acc<4>(acc[0]);
  zero_acc<4>(acc[1]);
  gemm2_acc<4>(acc, At, 512, w1, E_SZ, n0, E_SZ, lane);
  if (w2)
    gemm2_acc<4>(acc, At + 256, 512, w2, E_SZ, n0, E_SZ, lane);

#pragma unroll
  for (int rt=0; rt<2; rt++){
#pragma unroll
    for (int i=0;i<4;i++){
      int n = n0 + i*16 + row;
#pragma unroll
      for (int r=0;r<8;r++) G[(rt*16 + half*8 + r)*1024 + n] = acc[rt][i][r];
    }
  }
  __syncthreads();

  // gate pass: thread owns column e for 16 of the 32 rows
  const int e  = tid & 255;
  const int rb = (tid >> 8) * 16;
  float bi = b_ih[e]     + b_hh[e];
  float bf_= b_ih[e+256] + b_hh[e+256];
  float bg = b_ih[e+512] + b_hh[e+512];
  float bo = b_ih[e+768] + b_hh[e+768];
  float wi0=0.f, wi1=0.f, wf0=0.f, wf1=0.f, wg0=0.f, wg1=0.f, wo0=0.f, wo1=0.f;
  if (w_ih0){
    wi0=w_ih0[e*2];         wi1=w_ih0[e*2+1];
    wf0=w_ih0[(e+256)*2];   wf1=w_ih0[(e+256)*2+1];
    wg0=w_ih0[(e+512)*2];   wg1=w_ih0[(e+512)*2+1];
    wo0=w_ih0[(e+768)*2];   wo1=w_ih0[(e+768)*2+1];
  }
  for (int r = rb; r < rb+16; r++){
    int b = b0 + r;
    float x0=0.f, x1=0.f;
    if (w_ih0){ x0 = obs[b*(T_SZ*2) + t*2]; x1 = obs[b*(T_SZ*2) + t*2 + 1]; }
    float gi = G[r*1024 + e      ] + bi  + x0*wi0 + x1*wi1;
    float gf = G[r*1024 + e + 256] + bf_ + x0*wf0 + x1*wf1;
    float gg = G[r*1024 + e + 512] + bg  + x0*wg0 + x1*wg1;
    float go = G[r*1024 + e + 768] + bo  + x0*wo0 + x1*wo1;
    float cO = c_state[(size_t)b*E_SZ + e];
    float cN = sigm(gf)*cO + sigm(gi)*tanhf(gg);
    float hN = sigm(go)*tanhf(cN);
    c_state[(size_t)b*E_SZ + e] = cN;
    h_out[(size_t)b*E_SZ + e]  = (bf16)hN;
  }
}

// ---------------- fused per-mode head ----------------
// softmax over a size-1 axis == 1, so ctx == v; q/k are dead code.
// 256 threads = 8 waves; 16-row tile; NT<=8 keeps accumulator pressure low.
__global__ __launch_bounds__(256) void mode_head(
    const bf16* __restrict__ emb,
    const bf16* __restrict__ wb_v,   const float* __restrict__ in_proj_b,
    const bf16* __restrict__ wb_out, const float* __restrict__ out_b,
    const float* __restrict__ ln1_g, const float* __restrict__ ln1_b,
    const bf16* __restrict__ wb_f1,  const float* __restrict__ ffn_b1,
    const bf16* __restrict__ wb_f2,  const float* __restrict__ ffn_b2,
    const float* __restrict__ ln2_g, const float* __restrict__ ln2_b,
    const bf16* __restrict__ wb_d1,  const float* __restrict__ dec_b1,
    const bf16* __restrict__ wb_d2,  const float* __restrict__ dec_b2,
    const bf16* __restrict__ wb_s1,  const float* __restrict__ sc_b1,
    const float* __restrict__ sc_w2, const float* __restrict__ sc_b2,
    const float* __restrict__ obs,
    float* __restrict__ out_pred, float* __restrict__ out_score)
{
  extern __shared__ char smem[];
  bf16* T0 = (bf16*)smem;                   // 16x256 bf16 (8 KB)  : emb / ln outs
  bf16* T1 = (bf16*)(smem + 8192);          // 16x1024 bf16 (32 KB): ffn hidden
  bf16* T2 = (bf16*)(smem + 8192 + 32768);  // 16x512 bf16 (16 KB) : ctx / dec / sc
  float* F1 = (float*)T1;                   // 16x256 f32 staging (S1)
  float* F2 = (float*)T2;                   // 16x256 f32 staging (S3)

  const int b0 = blockIdx.x * 16;
  const int m  = blockIdx.y;
  const int tid = threadIdx.x, lane = tid & 31, wave = tid >> 5;

  // async-stage emb tile into T0 (LDS offset 0): 8KB = 512 x 16B chunks
  for (int i = tid; i < 512; i += 256)
    async_g2l_b128((unsigned)(i*16), (const char*)(emb + (size_t)b0*E_SZ) + i*16);
  wait_async0();
  __syncthreads();

  f32x8 acc[8];

  // S0: ctx = emb @ Wv^T + bv  -> T2 (ld 256)
  zero_acc<2>(acc);
  gemm_acc<2>(acc, T0, 256, wb_v + (size_t)m*256*256, 256, wave*32, 256, lane, 255);
  store_bf16<2,0>(acc, T2, 256, wave*32, in_proj_b + m*768 + 512, lane);
  __syncthreads();

  // S1: attn_out = ctx @ out_w^T + out_b -> F1 ; LN1 -> T0
  zero_acc<2>(acc);
  gemm_acc<2>(acc, T2, 256, wb_out + (size_t)m*256*256, 256, wave*32, 256, lane, 255);
  store_f32<2>(acc, F1, 256, wave*32, out_b + m*256, lane);
  __syncthreads();
  if (tid < 16) ln_row(F1 + tid*256, ln1_g, ln1_b, T0 + tid*256);
  __syncthreads();

  // S2: h = relu(T0 @ ffn_w1^T + b1) -> T1 (ld 1024)
  zero_acc<8>(acc);
  gemm_acc<8>(acc, T0, 256, wb_f1, 256, wave*128, 256, lane, 1023);
  store_bf16<8,1>(acc, T1, 1024, wave*128, ffn_b1, lane);
  __syncthreads();

  // S3: h2 = T1 @ ffn_w2^T + b2 -> F2 ; LN2 -> T0 (mm_emb)
  zero_acc<2>(acc);
  gemm_acc<2>(acc, T1, 1024, wb_f2, 1024, wave*32, 1024, lane, 255);
  store_f32<2>(acc, F2, 256, wave*32, ffn_b2, lane);
  __syncthreads();
  if (tid < 16) ln_row(F2 + tid*256, ln2_g, ln2_b, T0 + tid*256);
  __syncthreads();

  // S4: d = elu(T0 @ dec_w1^T + b) -> T2 (ld 512)
  zero_acc<4>(acc);
  gemm_acc<4>(acc, T0, 256, wb_d1, 256, wave*64, 256, lane, 511);
  store_bf16<4,2>(acc, T2, 512, wave*64, dec_b1, lane);
  __syncthreads();

  // S5: deltas = T2 @ dec_w2^T + b ; predictions = deltas + last_pos
  if (wave < 2) {
    zero_acc<1>(acc);
    gemm_acc<1>(acc, T2, 512, wb_d2, 512, wave*16, 512, lane, 23);
    int n = wave*16 + (lane & 15);
    if (n < 24) {
      int c = n & 1, fs = n >> 1;
#pragma unroll
      for (int r=0;r<8;r++){
        int b = b0 + (lane >> 4)*8 + r;
        float v = acc[0][r] + dec_b2[n] + obs[b*(T_SZ*2) + (T_SZ-1)*2 + c];
        out_pred[(((size_t)b*M_SZ + m)*FS_SZ + fs)*2 + c] = v;
      }
    }
  }
  __syncthreads();

  // S6: s = elu(T0 @ sc_w1^T + b) -> T2 (ld 128)
  zero_acc<1>(acc);
  gemm_acc<1>(acc, T0, 256, wb_s1, 256, wave*16, 256, lane, 127);
  store_bf16<1,2>(acc, T2, 128, wave*16, sc_b1, lane);
  __syncthreads();

  // S7: scores = s @ sc_w2^T + b
  if (tid < 16) {
    float s = sc_b2[0];
    for (int k=0;k<128;k++) s += (float)T2[tid*128 + k] * sc_w2[k];
    out_score[(size_t)(b0 + tid)*M_SZ + m] = s;
  }
}

extern "C" void kernel_launch(void* const* d_in, const int* in_sizes, int n_in,
                              void* d_out, int out_size, void* d_ws, size_t ws_size,
                              hipStream_t stream) {
  const float* obs       = (const float*)d_in[0];
  const float* w_ih0     = (const float*)d_in[1];
  const float* w_hh0     = (const float*)d_in[2];
  const float* b_ih0     = (const float*)d_in[3];
  const float* b_hh0     = (const float*)d_in[4];
  const float* w_ih1     = (const float*)d_in[5];
  const float* w_hh1     = (const float*)d_in[6];
  const float* b_ih1     = (const float*)d_in[7];
  const float* b_hh1     = (const float*)d_in[8];
  const float* in_proj_w = (const float*)d_in[9];
  const float* in_proj_b = (const float*)d_in[10];
  const float* out_w     = (const float*)d_in[11];
  const float* out_b     = (const float*)d_in[12];
  const float* ln1_g     = (const float*)d_in[13];
  const float* ln1_b     = (const float*)d_in[14];
  const float* ffn_w1    = (const float*)d_in[15];
  const float* ffn_b1    = (const float*)d_in[16];
  const float* ffn_w2    = (const float*)d_in[17];
  const float* ffn_b2    = (const float*)d_in[18];
  const float* ln2_g     = (const float*)d_in[19];
  const float* ln2_b     = (const float*)d_in[20];
  const float* dec_w1    = (const float*)d_in[21];
  const float* dec_b1    = (const float*)d_in[22];
  const float* dec_w2    = (const float*)d_in[23];
  const float* dec_b2    = (const float*)d_in[24];
  const float* sc_w1     = (const float*)d_in[25];
  const float* sc_b1     = (const float*)d_in[26];
  const float* sc_w2     = (const float*)d_in[27];
  const float* sc_b2     = (const float*)d_in[28];

  char* ws = (char*)d_ws;
  size_t off = 0;
  auto alloc = [&](size_t bytes)->char* {
    char* p = ws + off; off += (bytes + 255) & ~(size_t)255; return p;
  };
  bf16* wb_hh0 = (bf16*)alloc((size_t)1024*256*2);
  bf16* wb_ih1 = (bf16*)alloc((size_t)1024*256*2);
  bf16* wb_hh1 = (bf16*)alloc((size_t)1024*256*2);
  bf16* wb_v   = (bf16*)alloc((size_t)M_SZ*256*256*2);
  bf16* wb_out = (bf16*)alloc((size_t)M_SZ*256*256*2);
  bf16* wb_f1  = (bf16*)alloc((size_t)1024*256*2);
  bf16* wb_f2  = (bf16*)alloc((size_t)256*1024*2);
  bf16* wb_d1  = (bf16*)alloc((size_t)512*256*2);
  bf16* wb_d2  = (bf16*)alloc((size_t)24*512*2);
  bf16* wb_s1  = (bf16*)alloc((size_t)128*256*2);
  bf16* h0     = (bf16*)alloc((size_t)B_SZ*E_SZ*2);
  bf16* h1     = (bf16*)alloc((size_t)B_SZ*E_SZ*2);
  float* c0    = (float*)alloc((size_t)B_SZ*E_SZ*4);
  float* c1    = (float*)alloc((size_t)B_SZ*E_SZ*4);

  auto cvt = [&](const float* s, bf16* d, int n){
    cvt_bf16<<<(n+255)/256, 256, 0, stream>>>(s, d, n);
  };
  cvt(w_hh0, wb_hh0, 1024*256);
  cvt(w_ih1, wb_ih1, 1024*256);
  cvt(w_hh1, wb_hh1, 1024*256);
  cvt_vslice<<<(M_SZ*256*256)/256, 256, 0, stream>>>(in_proj_w, wb_v);
  cvt(out_w,  wb_out, M_SZ*256*256);
  cvt(ffn_w1, wb_f1, 1024*256);
  cvt(ffn_w2, wb_f2, 256*1024);
  cvt(dec_w1, wb_d1, 512*256);
  cvt(dec_w2, wb_d2, 24*512);
  cvt(sc_w1,  wb_s1, 128*256);

  hipMemsetAsync(h0, 0, (size_t)B_SZ*E_SZ*2, stream);
  hipMemsetAsync(h1, 0, (size_t)B_SZ*E_SZ*2, stream);
  hipMemsetAsync(c0, 0, (size_t)B_SZ*E_SZ*4, stream);
  hipMemsetAsync(c1, 0, (size_t)B_SZ*E_SZ*4, stream);

  dim3 lgrid(B_SZ/32);
  const size_t lstm_lds = 32*512*2 + 32*1024*4;      // 32KB A panel + 128KB gates
  for (int t = 0; t < T_SZ; t++) {
    // layer 0: g = h0 @ w_hh0^T + x_t @ w_ih0^T + biases ; updates (h0, c0)
    lstm_step<<<lgrid, 512, lstm_lds, stream>>>(h0, nullptr, wb_hh0, nullptr,
        b_ih0, b_hh0, w_ih0, obs, t, c0, h0);
    // layer 1: g = y0 @ w_ih1^T + h1 @ w_hh1^T + biases ; updates (h1, c1)
    lstm_step<<<lgrid, 512, lstm_lds, stream>>>(h0, h1, wb_ih1, wb_hh1,
        b_ih1, b_hh1, nullptr, obs, t, c1, h1);
  }

  float* out_pred  = (float*)d_out;
  float* out_score = out_pred + (size_t)B_SZ*M_SZ*FS_SZ*2;
  dim3 mgrid(B_SZ/16, M_SZ);
  const size_t head_lds = 8192 + 32768 + 16384;      // 56 KB
  mode_head<<<mgrid, 256, head_lds, stream>>>(h1,
      wb_v, in_proj_b, wb_out, out_b, ln1_g, ln1_b,
      wb_f1, ffn_b1, wb_f2, ffn_b2, ln2_g, ln2_b,
      wb_d1, dec_b1, wb_d2, dec_b2, wb_s1, sc_b1, sc_w2, sc_b2,
      obs, out_pred, out_score);
}
